// LinearVDO_44358422233577
// MI455X (gfx1250) — matrix operand order repaired and checked by reading, not compiled, not run on hardware
//
#include <hip/hip_runtime.h>

typedef _Float16 v16h __attribute__((ext_vector_type(16)));
typedef _Float16 v8h  __attribute__((ext_vector_type(8)));
typedef float    v8f  __attribute__((ext_vector_type(8)));

#define IN_F   768
#define OUT_F  768
#define BATCH  64
#define LDS_STRIDE 40   // 32 halves + 8 pad -> 80B row stride, conflict-free b128 reads
#define TILE_M 128
#define TILE_N 256

// ---------------------------------------------------------------------------
// Kernel 1: precision prep. Wh = f16(W); x2h = f16(x*x).
// ---------------------------------------------------------------------------
__global__ void vdo_prep(const float* __restrict__ W, const float* __restrict__ x,
                         _Float16* __restrict__ Wh, _Float16* __restrict__ x2h) {
    int i = blockIdx.x * blockDim.x + threadIdx.x;
    if (i < OUT_F * IN_F) Wh[i] = (_Float16)W[i];
    if (i < BATCH * IN_F) { float xv = x[i]; x2h[i] = (_Float16)(xv * xv); }
}

// ---------------------------------------------------------------------------
// Kernel 2: y_mean[b,o] = x_b . W_o + bias[o];  t1[b,o] = 2*alpha*sum_i (x*W)^2
// ---------------------------------------------------------------------------
__global__ void vdo_mean_t1(const float* __restrict__ x, const float* __restrict__ W,
                            const float* __restrict__ log_alpha,
                            const float* __restrict__ bias,
                            float* __restrict__ y_mean, float* __restrict__ t1) {
    int o = blockIdx.x * blockDim.x + threadIdx.x;   // 0..767
    int b = blockIdx.y;                              // 0..63
    float alpha = __expf(log_alpha[0]);
    const float4* xr = (const float4*)(x + (size_t)b * IN_F);
    const float4* wr = (const float4*)(W + (size_t)o * IN_F);
    float m = 0.f, t = 0.f;
    #pragma unroll 4
    for (int i = 0; i < IN_F / 4; ++i) {
        float4 xv = xr[i], wv = wr[i];
        float p0 = xv.x * wv.x, p1 = xv.y * wv.y, p2 = xv.z * wv.z, p3 = xv.w * wv.w;
        m += p0 + p1 + p2 + p3;
        t += p0*p0 + p1*p1 + p2*p2 + p3*p3;
    }
    y_mean[(size_t)b * OUT_F + o] = m + bias[o];
    t1[(size_t)b * OUT_F + o]     = 2.f * alpha * t;
}

// ---------------------------------------------------------------------------
// Kernel 3: y_var[b,o,p] = sum_i (x2[b,i]*W[o,i]) * W[p,i]  (+ t1[b,o] if o==p)
// Macro tile 128(M) x 256(N) per 256-thread block; 8 waves as 2M x 4N, each
// wave owns a 64x64 sub-tile = 16 accumulators -> 16 WMMAs per 16 fragment
// loads per K-chunk (1.0 ds_load_b128 per v_wmma, vs 1.5 for 64x32 tiles).
// ---------------------------------------------------------------------------
__global__ __launch_bounds__(256)
void vdo_var_gemm(const _Float16* __restrict__ Wh, const _Float16* __restrict__ x2h,
                  const float* __restrict__ t1, float* __restrict__ yvar) {
    __shared__ _Float16 sA[TILE_M * LDS_STRIDE];   // 10.0 KB
    __shared__ _Float16 sB[TILE_N * LDS_STRIDE];   // 20.0 KB
    __shared__ _Float16 sx2[IN_F];                 //  1.5 KB

    const int tid  = threadIdx.x;
    const int lane = tid & 31;
    const int wave = tid >> 5;
    const int wm   = wave & 1;     // wave M position (0..1) -> 64 rows
    const int wn   = wave >> 1;    // wave N position (0..3) -> 64 cols
    const int b      = blockIdx.z;
    const int o_base = blockIdx.x * TILE_M;
    const int p_base = blockIdx.y * TILE_N;

    // stage x^2 row (f16) for column scaling of the A tile
    for (int i = tid; i < IN_F; i += 256) sx2[i] = x2h[(size_t)b * IN_F + i];

    // fill mapping: A -> thread covers 16 halves of one of 128 rows;
    //               B -> thread covers all 32 halves of one of 256 rows.
    const int fr = tid >> 1;           // A tile row 0..127
    const int fs = (tid & 1) * 16;     // A half-offset within 32-wide K chunk
    const _Float16* gA = Wh + (size_t)(o_base + fr) * IN_F + fs;
    const _Float16* gB = Wh + (size_t)(p_base + tid) * IN_F;

    v8f acc[4][4];
    v8f zero = {};
    #pragma unroll
    for (int mt = 0; mt < 4; ++mt)
        #pragma unroll
        for (int nt = 0; nt < 4; ++nt) acc[mt][nt] = zero;

    for (int k0 = 0; k0 < IN_F; k0 += 32) {
        __syncthreads();  // prev iteration's fragment reads done; sx2 visible
        // A tile: column-scaled by x^2
        v8h s0 = *(const v8h*)&sx2[k0 + fs];
        v8h s1 = *(const v8h*)&sx2[k0 + fs + 8];
        v8h a0 = *(const v8h*)(gA + k0);
        v8h a1 = *(const v8h*)(gA + k0 + 8);
        a0 *= s0; a1 *= s1;
        // B tile: raw W rows (x^2 folded into A only: (x2*Wo)*Wp)
        v8h b0 = *(const v8h*)(gB + k0);
        v8h b1 = *(const v8h*)(gB + k0 + 8);
        v8h b2 = *(const v8h*)(gB + k0 + 16);
        v8h b3 = *(const v8h*)(gB + k0 + 24);
        *(v8h*)&sA[fr * LDS_STRIDE + fs]      = a0;
        *(v8h*)&sA[fr * LDS_STRIDE + fs + 8]  = a1;
        *(v8h*)&sB[tid * LDS_STRIDE]          = b0;
        *(v8h*)&sB[tid * LDS_STRIDE + 8]      = b1;
        *(v8h*)&sB[tid * LDS_STRIDE + 16]     = b2;
        *(v8h*)&sB[tid * LDS_STRIDE + 24]     = b3;
        __syncthreads();

        // A fragments: lanes 0-15 hold K{0..7,16..23}, lanes 16-31 K{8..15,24..31}
        v16h af[4], bf[4];
        const int aoff = (lane < 16) ? 0 : 8;
        #pragma unroll
        for (int mt = 0; mt < 4; ++mt) {
            int row = wm * 64 + mt * 16 + (lane & 15);
            v8h lo = *(const v8h*)&sA[row * LDS_STRIDE + aoff];
            v8h hi = *(const v8h*)&sA[row * LDS_STRIDE + aoff + 16];
            #pragma unroll
            for (int j = 0; j < 8; ++j) { af[mt][j] = lo[j]; af[mt][j + 8] = hi[j]; }
        }
        // B fragments: lanes 0-15 hold K{0..7,16..23} of column n, lanes 16-31 K{8..15,24..31}
        const int boff = (lane < 16) ? 0 : 8;
        #pragma unroll
        for (int nt = 0; nt < 4; ++nt) {
            int row = wn * 64 + nt * 16 + (lane & 15);
            v8h lo = *(const v8h*)&sB[row * LDS_STRIDE + boff];
            v8h hi = *(const v8h*)&sB[row * LDS_STRIDE + boff + 16];
            #pragma unroll
            for (int j = 0; j < 8; ++j) { bf[nt][j] = lo[j]; bf[nt][j + 8] = hi[j]; }
        }
        #pragma unroll
        for (int mt = 0; mt < 4; ++mt)
            #pragma unroll
            for (int nt = 0; nt < 4; ++nt)
                acc[mt][nt] = __builtin_amdgcn_wmma_f32_16x16x32_f16(
                    false, af[mt], false, bf[nt], (short)0, acc[mt][nt], false, false);
    }

    // Epilogue. C layout: VGPR r, lanes 0-15 -> M=r,N=lane ; lanes 16-31 ->
    // M=r+8, N=lane-16. Diagonal handling only in blocks whose o/p ranges
    // intersect (uniform scalar branch); those use an unconditional coalesced
    // t1 load per N-subtile + branchless cndmask fold.
    float* outb = yvar + (size_t)b * OUT_F * OUT_F;
    const int m_half = (lane >> 4) * 8;
    const int n_loc  = lane & 15;
    const bool has_diag = (o_base < p_base + TILE_N) && (p_base < o_base + TILE_M);

    if (has_diag) {
        const float* t1row = t1 + (size_t)b * OUT_F;
        #pragma unroll
        for (int nt = 0; nt < 4; ++nt) {
            const int p = p_base + wn * 64 + nt * 16 + n_loc;
            const float t1p = t1row[p];
            #pragma unroll
            for (int mt = 0; mt < 4; ++mt) {
                const int o0 = o_base + wm * 64 + mt * 16 + m_half;
                float* outp = outb + (size_t)o0 * OUT_F + p;
                #pragma unroll
                for (int r = 0; r < 8; ++r) {
                    float v = acc[mt][nt][r];
                    v += (o0 + r == p) ? t1p : 0.0f;   // v_cmp + v_cndmask
                    outp[(size_t)r * OUT_F] = v;
                }
            }
        }
    } else {
        #pragma unroll
        for (int nt = 0; nt < 4; ++nt) {
            const int p = p_base + wn * 64 + nt * 16 + n_loc;
            #pragma unroll
            for (int mt = 0; mt < 4; ++mt) {
                const int o0 = o_base + wm * 64 + mt * 16 + m_half;
                float* outp = outb + (size_t)o0 * OUT_F + p;
                #pragma unroll
                for (int r = 0; r < 8; ++r)
                    outp[(size_t)r * OUT_F] = acc[mt][nt][r];
            }
        }
    }
}

// ---------------------------------------------------------------------------
extern "C" void kernel_launch(void* const* d_in, const int* in_sizes, int n_in,
                              void* d_out, int out_size, void* d_ws, size_t ws_size,
                              hipStream_t stream) {
    const float* x    = (const float*)d_in[0];   // [64,768]
    const float* W    = (const float*)d_in[1];   // [768,768]
    const float* la   = (const float*)d_in[2];   // [1,1]
    const float* bias = (const float*)d_in[3];   // [1,768]

    float* out    = (float*)d_out;
    float* y_mean = out;                          // 64*768
    float* yvar   = out + (size_t)BATCH * OUT_F;  // 64*768*768

    // workspace layout (all 16B aligned): Wh | x2h | t1  (~1.5 MB total)
    _Float16* Wh  = (_Float16*)d_ws;
    _Float16* x2h = (_Float16*)((char*)d_ws + (size_t)OUT_F * IN_F * 2);
    float*    t1  = (float*)((char*)d_ws + (size_t)OUT_F * IN_F * 2
                                         + (size_t)BATCH * IN_F * 2);

    vdo_prep<<<(OUT_F * IN_F + 255) / 256, 256, 0, stream>>>(W, x, Wh, x2h);
    vdo_mean_t1<<<dim3(OUT_F / 256, BATCH), 256, 0, stream>>>(x, W, la, bias, y_mean, t1);
    vdo_var_gemm<<<dim3(OUT_F / TILE_M, OUT_F / TILE_N, BATCH), 256, 0, stream>>>(
        Wh, x2h, t1, yvar);
}
